// ClusteredAttention_53815940219063
// MI455X (gfx1250) — matrix operand order repaired and checked by reading, not compiled, not run on hardware
//
#include <hip/hip_runtime.h>
#include <math.h>

typedef _Float16 half_t;
typedef __attribute__((ext_vector_type(16))) _Float16 v16h;
typedef __attribute__((ext_vector_type(8)))  _Float16 v8h;
typedef __attribute__((ext_vector_type(8)))  float    v8f;

#define B_ 4
#define L_ 4096
#define D_ 512
#define NC_ 32
#define DCHUNK 128
#define NTILE_O (DCHUNK/16)     // 8 output col-tiles per wave
#define WPB 4                   // waves (M-tiles) per workgroup
#define ROWS (WPB*16)           // 64 rows per workgroup
#define MTGROUPS (L_/ROWS)      // worst-case row-tile groups per cluster (64)
#define KSLICE 128              // staged K depth per barrier round
#define CH 16                   // L_/256 elements per thread in binning

// ---------------------------------------------------------------------------
// Kernel 1: deterministic per-(batch,cluster) binning via stable chunked scan.
// ---------------------------------------------------------------------------
__global__ __launch_bounds__(256) void ca_bin_kernel(
    const long long* __restrict__ labels,
    int* __restrict__ counts, int* __restrict__ offsets, int* __restrict__ idxbuf)
{
    __shared__ int scanbuf[256];
    __shared__ int base_sh;
    const int b   = blockIdx.x;
    const int tid = threadIdx.x;
    if (tid == 0) base_sh = 0;
    __syncthreads();

    for (int c = 0; c < NC_; ++c) {
        int mylist[CH];
        int cnt = 0;
        for (int k = 0; k < CH; ++k) {
            int i = tid * CH + k;
            if ((int)labels[(size_t)b * L_ + i] == c) mylist[cnt++] = i;
        }
        scanbuf[tid] = cnt;
        __syncthreads();
        for (int off = 1; off < 256; off <<= 1) {
            int vcur = scanbuf[tid];
            int vadd = (tid >= off) ? scanbuf[tid - off] : 0;
            __syncthreads();
            scanbuf[tid] = vcur + vadd;
            __syncthreads();
        }
        const int incl  = scanbuf[tid];
        const int total = scanbuf[255];
        const int excl  = incl - cnt;
        const int off0  = base_sh;
        for (int k = 0; k < cnt; ++k)
            idxbuf[(size_t)b * L_ + off0 + excl + k] = mylist[k];
        if (tid == 0) { counts[b * NC_ + c] = total; offsets[b * NC_ + c] = off0; }
        __syncthreads();
        if (tid == 0) base_sh = off0 + total;
        __syncthreads();
    }
}

// ---------------------------------------------------------------------------
// Kernel 2: per-cluster flash attention. One workgroup = 4 waves = 4 adjacent
// 16-row M-tiles of one cluster, one 128-col D-chunk. Staging is row-per-
// thread-group with unrolled float4 loads so many global_load_b128 are in
// flight; WMMA f32_16x16x32_f16 for QK^T and PV.
// ---------------------------------------------------------------------------
__global__ __launch_bounds__(WPB*32) void ca_attn_kernel(
    const float* __restrict__ q, const float* __restrict__ kmat,
    const float* __restrict__ vmat,
    const int* __restrict__ counts, const int* __restrict__ offsets,
    const int* __restrict__ idxbuf, float* __restrict__ out)
{
    __shared__ __align__(16) half_t Qs[ROWS][D_];       // 64 KB
    __shared__ __align__(16) half_t Ks[32][KSLICE];     //  8 KB
    __shared__ __align__(16) half_t VsT[DCHUNK][32];    //  8 KB (col-major V)
    __shared__ __align__(16) half_t Ps[WPB][16][32];    //  4 KB
    __shared__ int ridx[ROWS];
    __shared__ int cidx[32];

    const int t   = blockIdx.x;
    const int mtg = t & (MTGROUPS - 1);                 // row-tile group
    const int c   = (t / MTGROUPS) & (NC_ - 1);
    const int b   = t / (MTGROUPS * NC_);
    const int dbase = blockIdx.y * DCHUNK;

    const int count = counts[b * NC_ + c];
    if (mtg * ROWS >= count) return;                    // workgroup-uniform exit
    const int off = offsets[b * NC_ + c];
    const int* ilist = idxbuf + (size_t)b * L_ + off;

    const int tid  = threadIdx.x;
    const int w    = tid >> 5;                          // wave id: which M-tile
    const int lane = tid & 31;
    const int lo = lane & 15;
    const int hi = lane >> 4;
    const int rowbase = mtg * ROWS + w * 16;            // this wave's first row

    if (tid < ROWS) {
        int rg = mtg * ROWS + tid;
        ridx[tid] = ilist[rg < count ? rg : count - 1];
    }
    __syncthreads();

    // ---- stage Q rows (scaled, f16): 2 threads per row, unrolled float4 ----
    const float scale = 0.044194173824159216f;          // 1/sqrt(512)
    {
        const int qrow  = tid >> 1;                     // 0..63
        const int qcol0 = (tid & 1) * (D_ / 2);         // 0 or 256
        const float* __restrict__ qsrc =
            &q[((size_t)b * L_ + ridx[qrow]) * D_ + qcol0];
        #pragma unroll 8
        for (int u = 0; u < (D_ / 2) / 4; ++u) {        // 64 float4 per thread
            const float4 qv = *(const float4*)&qsrc[u * 4];
            const int col = qcol0 + u * 4;
            Qs[qrow][col]     = (half_t)(qv.x * scale);
            Qs[qrow][col + 1] = (half_t)(qv.y * scale);
            Qs[qrow][col + 2] = (half_t)(qv.z * scale);
            Qs[qrow][col + 3] = (half_t)(qv.w * scale);
        }
    }
    __syncthreads();

    float mrow[8], lrow[8];
    v8f o[NTILE_O];
    for (int r = 0; r < 8; ++r) { mrow[r] = -__builtin_inff(); lrow[r] = 0.f; }
    for (int n = 0; n < NTILE_O; ++n)
        for (int r = 0; r < 8; ++r) o[n][r] = 0.f;

    // staging assignment: 4 threads per gathered member row
    const int smem  = tid >> 2;                         // member 0..31
    const int scol0 = (tid & 3) * 32;                   // 32-col span

    const int ntiles = (count + 31) / 32;
    for (int jt = 0; jt < ntiles; ++jt) {
        const int j0 = jt * 32;
        if (tid < 32) {
            int j = j0 + tid;
            cidx[tid] = ilist[j < count ? j : count - 1];
        }
        __syncthreads();

        // per-tile gathered row bases (computed once, reused for K and V)
        const size_t memrow = (size_t)b * L_ + cidx[smem];
        const float* __restrict__ krow = &kmat[memrow * D_ + scol0];
        const float* __restrict__ vrow = &vmat[memrow * D_ + dbase + scol0];

        // ---- stage V^T chunk: VsT[col][member], 8 float4 in flight ----
        #pragma unroll
        for (int u = 0; u < 8; ++u) {
            const float4 vv = *(const float4*)&vrow[u * 4];
            const int col = scol0 + u * 4;
            VsT[col][smem]     = (half_t)vv.x;
            VsT[col + 1][smem] = (half_t)vv.y;
            VsT[col + 2][smem] = (half_t)vv.z;
            VsT[col + 3][smem] = (half_t)vv.w;
        }

        // ---- S = Q * K^T: stage K 128-deep, 8 WMMAs per barrier round ----
        v8f s0, s1;
        for (int r = 0; r < 8; ++r) { s0[r] = 0.f; s1[r] = 0.f; }
        for (int kk0 = 0; kk0 < D_; kk0 += KSLICE) {
            __syncthreads();
            #pragma unroll
            for (int u = 0; u < 8; ++u) {               // Ks[member][kcol]
                const float4 kv = *(const float4*)&krow[kk0 + u * 4];
                const int col = scol0 + u * 4;
                Ks[smem][col]     = (half_t)kv.x;
                Ks[smem][col + 1] = (half_t)kv.y;
                Ks[smem][col + 2] = (half_t)kv.z;
                Ks[smem][col + 3] = (half_t)kv.w;
            }
            __syncthreads();
            #pragma unroll
            for (int kks = 0; kks < KSLICE; kks += 32) {
                const int kk = kk0 + kks;
                v16h a;
                {
                    const v8h a0 = *(const v8h*)&Qs[w * 16 + lo][kk + 16 * hi];
                    const v8h a1 = *(const v8h*)&Qs[w * 16 + lo][kk + 16 * hi + 8];
                    for (int i = 0; i < 8; ++i) { a[i] = a0[i]; a[8 + i] = a1[i]; }
                }
                v16h b0, b1;
                {
                    const v8h x0 = *(const v8h*)&Ks[lo][kks + 16 * hi];
                    const v8h x1 = *(const v8h*)&Ks[lo][kks + 16 * hi + 8];
                    const v8h y0 = *(const v8h*)&Ks[16 + lo][kks + 16 * hi];
                    const v8h y1 = *(const v8h*)&Ks[16 + lo][kks + 16 * hi + 8];
                    for (int i = 0; i < 8; ++i) {
                        b0[i] = x0[i]; b0[8 + i] = x1[i];
                        b1[i] = y0[i]; b1[8 + i] = y1[i];
                    }
                }
                s0 = __builtin_amdgcn_wmma_f32_16x16x32_f16(false, a, false, b0, (short)0, s0, false, false);
                s1 = __builtin_amdgcn_wmma_f32_16x16x32_f16(false, a, false, b1, (short)0, s1, false, false);
            }
        }

        // ---- mask out-of-range columns ----
        if (j0 + lo >= count)      for (int r = 0; r < 8; ++r) s0[r] = -__builtin_inff();
        if (j0 + 16 + lo >= count) for (int r = 0; r < 8; ++r) s1[r] = -__builtin_inff();

        // ---- online softmax (rows r+8*hi live across the 16-lane half) ----
        float corr[8], rowred[8];
        for (int r = 0; r < 8; ++r) rowred[r] = fmaxf(s0[r], s1[r]);
        for (int m = 1; m < 16; m <<= 1)
            for (int r = 0; r < 8; ++r)
                rowred[r] = fmaxf(rowred[r], __shfl_xor(rowred[r], m, 32));
        float p0[8], p1[8];
        for (int r = 0; r < 8; ++r) {
            float mn = fmaxf(mrow[r], rowred[r]);
            corr[r]  = __expf(mrow[r] - mn);
            mrow[r]  = mn;
            p0[r] = __expf(s0[r] - mn);
            p1[r] = __expf(s1[r] - mn);
            rowred[r] = p0[r] + p1[r];
        }
        for (int m = 1; m < 16; m <<= 1)
            for (int r = 0; r < 8; ++r)
                rowred[r] += __shfl_xor(rowred[r], m, 32);
        for (int r = 0; r < 8; ++r) lrow[r] = lrow[r] * corr[r] + rowred[r];

        // ---- transpose P through LDS into A-fragment layout (per-wave slice) ----
        __syncthreads();
        for (int r = 0; r < 8; ++r) {
            Ps[w][r + 8 * hi][lo]      = (half_t)p0[r];
            Ps[w][r + 8 * hi][16 + lo] = (half_t)p1[r];
        }
        __syncthreads();

        for (int n = 0; n < NTILE_O; ++n)
            for (int r = 0; r < 8; ++r) o[n][r] *= corr[r];

        v16h ap;
        {
            const v8h a0 = *(const v8h*)&Ps[w][lo][16 * hi];
            const v8h a1 = *(const v8h*)&Ps[w][lo][16 * hi + 8];
            for (int i = 0; i < 8; ++i) { ap[i] = a0[i]; ap[8 + i] = a1[i]; }
        }
        #pragma unroll
        for (int n = 0; n < NTILE_O; ++n) {
            v16h bv;
            const v8h x0 = *(const v8h*)&VsT[n * 16 + lo][16 * hi];
            const v8h x1 = *(const v8h*)&VsT[n * 16 + lo][16 * hi + 8];
            for (int i = 0; i < 8; ++i) { bv[i] = x0[i]; bv[8 + i] = x1[i]; }
            o[n] = __builtin_amdgcn_wmma_f32_16x16x32_f16(false, ap, false, bv, (short)0, o[n], false, false);
        }
        __syncthreads();   // before next tile overwrites cidx/VsT/Ks
    }

    // ---- normalize + scatter output rows ----
    float invl[8];
    for (int r = 0; r < 8; ++r) invl[r] = 1.0f / lrow[r];
    for (int r = 0; r < 8; ++r) {
        if (rowbase + r + 8 * hi < count) {
            const int orow = ridx[w * 16 + r + 8 * hi];
            float* op = &out[((size_t)b * L_ + orow) * D_ + dbase + lo];
            for (int n = 0; n < NTILE_O; ++n)
                op[n * 16] = o[n][r] * invl[r];
        }
    }
}

// ---------------------------------------------------------------------------
extern "C" void kernel_launch(void* const* d_in, const int* in_sizes, int n_in,
                              void* d_out, int out_size, void* d_ws, size_t ws_size,
                              hipStream_t stream)
{
    const float*     q      = (const float*)d_in[0];
    const float*     k      = (const float*)d_in[1];
    const float*     v      = (const float*)d_in[2];
    const long long* labels = (const long long*)d_in[3];
    float*           out    = (float*)d_out;

    int* counts  = (int*)d_ws;                 // B*NC
    int* offsets = counts + B_ * NC_;          // B*NC
    int* idxbuf  = offsets + B_ * NC_;         // B*L

    ca_bin_kernel<<<B_, 256, 0, stream>>>(labels, counts, offsets, idxbuf);

    dim3 grid(B_ * NC_ * MTGROUPS, D_ / DCHUNK);
    ca_attn_kernel<<<grid, WPB * 32, 0, stream>>>(q, k, v, counts, offsets, idxbuf, out);
}
